// GrokOneDecoderBlock_46617575031311
// MI455X (gfx1250) — matrix operand order repaired and checked by reading, compiled, not run on hardware
//
#include <hip/hip_runtime.h>
#include <hip/hip_bf16.h>
#include <math.h>
#include <stdint.h>

typedef __attribute__((ext_vector_type(16))) __bf16 v16bf;
typedef __attribute__((ext_vector_type(8)))  __bf16 v8bf;
typedef __attribute__((ext_vector_type(8)))  float  v8f;
typedef __attribute__((ext_vector_type(4)))  uint32_t u32x4;
typedef __attribute__((ext_vector_type(8)))  uint32_t u32x8;

#define DEV __device__ __forceinline__

// ---------------- problem constants ----------------
constexpr int B_  = 2;
constexpr int S_  = 2048;
constexpr int D_  = 1024;
constexpr int H_  = 16;
constexpr int KH  = 64;          // head dim
constexpr int E_  = 8;
constexpr int F_  = 4096;
constexpr int MT  = B_ * S_;     // 4096 tokens

// GEMM tiling
constexpr int BK  = 64;          // K per LDS stage
constexpr int LDP = BK + 8;      // padded LDS row pitch (elems): 144B -> conflict-free

// ---------------- workspace layout (bytes) ----------------
constexpr size_t MB       = 1ull << 20;
constexpr size_t OFF_N1   = 0;          // bf16 [MT,D]   8MB
constexpr size_t OFF_Q    = 8  * MB;    // bf16 [B,H,S,64]
constexpr size_t OFF_K    = 16 * MB;
constexpr size_t OFF_V    = 24 * MB;
constexpr size_t OFF_O    = 32 * MB;
constexpr size_t OFF_XRES = 40 * MB;    // f32  [MT,D]  16MB
constexpr size_t OFF_N2   = 56 * MB;    // bf16 [MT,D]   8MB
constexpr size_t OFF_W    = 64 * MB;    // f32  [MT,E] 128KB
constexpr size_t OFF_HV   = 65 * MB;    // bf16 [MT,F]  32MB
constexpr size_t OFF_WQ   = 97 * MB;    // bf16 weight mirrors
constexpr size_t OFF_WK   = 99 * MB;
constexpr size_t OFF_WV   = 101 * MB;
constexpr size_t OFF_WO   = 103 * MB;
constexpr size_t OFF_WEIN = 105 * MB;   // 64MB
constexpr size_t OFF_WEV  = 169 * MB;   // 64MB
constexpr size_t OFF_WEO  = 233 * MB;   // 64MB  (end = 297MB)

// ---------------- WMMA helpers ----------------
DEV v8f vzero8() {
  v8f z;
#pragma unroll
  for (int i = 0; i < 8; ++i) z[i] = 0.0f;
  return z;
}

DEV v8f wmma_bf16(v16bf a, v16bf b, v8f c) {
  return __builtin_amdgcn_wmma_f32_16x16x32_bf16(false, a, false, b, (short)0, c,
                                                 false, false);
}

DEV v16bf cat8(v8bf lo, v8bf hb) {
  v16bf a;
#pragma unroll
  for (int i = 0; i < 8; ++i) { a[i] = lo[i]; a[8 + i] = hb[i]; }
  return a;
}

// A-fragment (16x32 bf16): low lanes hold K 0..15, high lanes 16..31.
DEV v16bf load_a_frag(const __bf16* A, int ld, int row, int kk, int lane) {
  int hi = (lane >> 4) & 1;
  const __bf16* p = A + (size_t)row * ld + kk + hi * 16;
  return cat8(*(const v8bf*)p, *(const v8bf*)(p + 8));
}

// B-fragment (32x16): lane holds column (lane&15); low lanes K=kk..+15, high +16..+31.
DEV v16bf load_b_frag(const __bf16* W, int ld, int col, int kk, int lane) {
  int hi = (lane >> 4) & 1;
  const __bf16* p = W + (size_t)col * ld + kk + hi * 16;
  return cat8(*(const v8bf*)p, *(const v8bf*)(p + 8));
}

// ---------------- TDM: 2D tile (64 x rows) global -> padded LDS ----------------
// D# per CDNA5 ISA: group0 (4 SGPRs) + group1 (8 SGPRs).
// pad_interval=4 (32 DWORDs = one 128B row), pad_amount=3 (4 DWORDs = 16B) -> LDP pitch.
DEV void tdm_load_tile(unsigned lds_off, const void* gaddr, unsigned rows,
                       unsigned stride_elems) {
  uint64_t ga = (uint64_t)(uintptr_t)gaddr;
  u32x4 g0;
  g0[0] = 1u;                                            // count=1, user desc
  g0[1] = lds_off;                                       // lds_addr (bytes)
  g0[2] = (uint32_t)ga;                                  // global_addr[31:0]
  g0[3] = (uint32_t)((ga >> 32) & 0x01FFFFFFu) | (2u << 30);  // addr[56:32] | type=2
  u32x8 g1;
  g1[0] = (1u << 16) | (1u << 20) | (4u << 22) | (3u << 25);  // 2B elems, pad on
  g1[1] = (BK & 0xFFFFu) << 16;                          // tensor_dim0 lo16 (=64)
  g1[2] = ((rows & 0xFFFFu) << 16);                      // dim0 hi=0 | tensor_dim1 lo16
  g1[3] = (BK << 16);                                    // dim1 hi=0 | tile_dim0=64
  g1[4] = rows & 0xFFFFu;                                // tile_dim1 | tile_dim2=0
  g1[5] = stride_elems;                                  // tensor_dim0_stride lo32
  g1[6] = 0;                                             // stride hi | dim1_stride lo
  g1[7] = 0;
  asm volatile("tensor_load_to_lds %0, %1" :: "s"(g0), "s"(g1) : "memory");
}

// Batch all fragment loads before the 8 WMMAs so DS latency overlaps matrix issue.
DEV void compute_stage(const __bf16* As, const __bf16* Bs, int wave, int lane,
                       v8f acc[4]) {
  int col = lane & 15;
  v16bf a0 = load_a_frag(As, LDP, wave * 16 + col, 0, lane);
  v16bf a1 = load_a_frag(As, LDP, wave * 16 + col, 32, lane);
  v16bf b[8];
#pragma unroll
  for (int kt = 0; kt < 2; ++kt)
#pragma unroll
    for (int j = 0; j < 4; ++j)
      b[kt * 4 + j] = load_b_frag(Bs, LDP, j * 16 + col, kt * 32, lane);
#pragma unroll
  for (int j = 0; j < 4; ++j) acc[j] = wmma_bf16(a0, b[j], acc[j]);
#pragma unroll
  for (int j = 0; j < 4; ++j) acc[j] = wmma_bf16(a1, b[4 + j], acc[j]);
}

DEV void compute_stage_dual(const __bf16* As, const __bf16* Bs0, const __bf16* Bs1,
                            int wave, int lane, v8f acc0[4], v8f acc1[4]) {
  int col = lane & 15;
#pragma unroll
  for (int kt = 0; kt < 2; ++kt) {
    v16bf a = load_a_frag(As, LDP, wave * 16 + col, kt * 32, lane);
    v16bf b0[4], b1[4];
#pragma unroll
    for (int j = 0; j < 4; ++j) {
      b0[j] = load_b_frag(Bs0, LDP, j * 16 + col, kt * 32, lane);
      b1[j] = load_b_frag(Bs1, LDP, j * 16 + col, kt * 32, lane);
    }
#pragma unroll
    for (int j = 0; j < 4; ++j) {
      acc0[j] = wmma_bf16(a, b0[j], acc0[j]);
      acc1[j] = wmma_bf16(a, b1[j], acc1[j]);
    }
  }
}

// Double-buffered TDM GEMM main loop: DMA of stage s+1 overlaps WMMA of stage s.
// Stage s: wave0 waits TENSORcnt (tiles s landed) -> barrier (also: everyone done
// computing s-1 from the other buffer) -> wave0 issues tiles s+1 into that buffer
// -> all waves compute stage s.  One barrier per stage.
DEV void gemm_tdm_loop(const __bf16* A, const __bf16* W, int Kd, int m0, int n0,
                       int wave, int lane, __bf16* As0, __bf16* As1,
                       __bf16* Bs0, __bf16* Bs1, v8f acc[4]) {
  unsigned a_off[2] = {(unsigned)(uintptr_t)As0, (unsigned)(uintptr_t)As1};
  unsigned b_off[2] = {(unsigned)(uintptr_t)Bs0, (unsigned)(uintptr_t)Bs1};
  const __bf16* Ab = A + (size_t)m0 * Kd;
  const __bf16* Wb = W + (size_t)n0 * Kd;
  if (wave == 0) {
    tdm_load_tile(a_off[0], Ab, 128, Kd);
    tdm_load_tile(b_off[0], Wb, 64, Kd);
  }
  int nst = Kd / BK;
  for (int s = 0; s < nst; ++s) {
    int cur = s & 1;
    if (wave == 0) __builtin_amdgcn_s_wait_tensorcnt(0);
    __syncthreads();
    if (wave == 0 && s + 1 < nst) {
      tdm_load_tile(a_off[cur ^ 1], Ab + (s + 1) * BK, 128, Kd);
      tdm_load_tile(b_off[cur ^ 1], Wb + (s + 1) * BK, 64, Kd);
    }
    compute_stage(cur ? As1 : As0, cur ? Bs1 : Bs0, wave, lane, acc);
  }
}

// ---------------- f32 -> bf16 weight conversion ----------------
__global__ __launch_bounds__(256) void cvt_bf16_kernel(const float* __restrict__ s,
                                                       __bf16* __restrict__ d,
                                                       long n) {
  long i = ((long)blockIdx.x * 256 + threadIdx.x) * 4;
  long stride = (long)gridDim.x * 1024;
  for (; i < n; i += stride) {
    float4 v = *(const float4*)(s + i);
    __builtin_prefetch(s + i + stride, 0, 1);
    d[i + 0] = (__bf16)v.x;
    d[i + 1] = (__bf16)v.y;
    d[i + 2] = (__bf16)v.z;
    d[i + 3] = (__bf16)v.w;
  }
}

// ---------------- RMSNorm (f32 in -> bf16 out) ----------------
__global__ __launch_bounds__(256) void rmsnorm_kernel(const float* __restrict__ x,
                                                      const float* __restrict__ scale,
                                                      __bf16* __restrict__ out) {
  __shared__ float red[256];
  int row = blockIdx.x, tid = threadIdx.x;
  const float4* xv = (const float4*)(x + (size_t)row * D_);
  float4 v = xv[tid];
  red[tid] = v.x * v.x + v.y * v.y + v.z * v.z + v.w * v.w;
  __syncthreads();
  for (int s = 128; s > 0; s >>= 1) {
    if (tid < s) red[tid] += red[tid + s];
    __syncthreads();
  }
  float rs = rsqrtf(red[0] * (1.0f / D_) + 1e-5f);
  float4 sc = ((const float4*)scale)[tid];
  size_t base = (size_t)row * D_ + tid * 4;
  out[base + 0] = (__bf16)(sc.x * v.x * rs);
  out[base + 1] = (__bf16)(sc.y * v.y * rs);
  out[base + 2] = (__bf16)(sc.z * v.z * rs);
  out[base + 3] = (__bf16)(sc.w * v.w * rs);
}

// ---------------- QKV projection: [MT,D] x W^T -> [B,H,S,64] bf16 ----------
__global__ __launch_bounds__(256) void gemm_qkv_kernel(const __bf16* __restrict__ A,
                                                       const __bf16* __restrict__ W,
                                                       __bf16* __restrict__ dst) {
  __shared__ __bf16 As[2][128 * LDP];
  __shared__ __bf16 Bs[2][64 * LDP];
  int tid = threadIdx.x, lane = tid & 31, wave = tid >> 5;
  int col = lane & 15, hi = (lane >> 4) & 1;
  int m0 = blockIdx.y * 128, n0 = blockIdx.x * 64;
  v8f acc[4];
#pragma unroll
  for (int j = 0; j < 4; ++j) acc[j] = vzero8();
  gemm_tdm_loop(A, W, D_, m0, n0, wave, lane, As[0], As[1], Bs[0], Bs[1], acc);
#pragma unroll
  for (int j = 0; j < 4; ++j)
#pragma unroll
    for (int r = 0; r < 8; ++r) {
      int m = m0 + wave * 16 + r + 8 * hi;   // token
      int n = n0 + j * 16 + col;             // h*64 + kd
      int bI = m / S_, sI = m % S_, hI = n >> 6, kd = n & 63;
      dst[(((size_t)bI * H_ + hI) * S_ + sI) * KH + kd] = (__bf16)acc[j][r];
    }
}

// ---------------- flash attention: per (b,h), 128 queries / block ----------
__global__ __launch_bounds__(256) void attn_kernel(const __bf16* __restrict__ Q,
                                                   const __bf16* __restrict__ Kb,
                                                   const __bf16* __restrict__ Vb,
                                                   __bf16* __restrict__ O) {
  __shared__ __bf16 ks[32][64];     // K chunk [key][dim] (async-copied)
  __shared__ __bf16 vt[64][32];     // V chunk transposed [dim][key]
  __shared__ __bf16 pw[8][16][32];  // per-wave probabilities [row][key]
  int tid = threadIdx.x;
  int lane = tid & 31, wave = tid >> 5;
  int col = lane & 15, hi = (lane >> 4) & 1;
  int bh = blockIdx.y;
  int b = bh / H_, h = bh % H_;
  size_t base = (size_t)bh * S_ * KH;
  int q0 = blockIdx.x * 128;
  int qm0 = q0 + wave * 16;

  v16bf aq0 = load_a_frag(Q + base, KH, qm0 + col, 0, lane);
  v16bf aq1 = load_a_frag(Q + base, KH, qm0 + col, 32, lane);

  float mrun[8], lrun[8];
  v8f acc[4];
#pragma unroll
  for (int r = 0; r < 8; ++r) { mrun[r] = -INFINITY; lrun[r] = 0.0f; }
#pragma unroll
  for (int j = 0; j < 4; ++j) acc[j] = vzero8();

  unsigned ks_off = (unsigned)(uintptr_t)&ks[0][0];
  int nchunk = (q0 + 128) / 32;
  for (int c = 0; c < nchunk; ++c) {
    int t = c * 32;
    __syncthreads();
    // --- K chunk: async global->LDS copy (8B per lane, 2 issues/thread) ---
    {
      const char* ksrc = (const char*)(Kb + base + (size_t)t * KH);
      unsigned o = (unsigned)tid * 8u;
#pragma unroll
      for (int i = 0; i < 2; ++i) {
        uint64_t ga = (uint64_t)(uintptr_t)(ksrc + o);
        asm volatile("global_load_async_to_lds_b64 %0, %1, off"
                     :: "v"(ks_off + o), "v"(ga) : "memory");
        o += 2048u;
      }
    }
    // --- V chunk: transpose into LDS with regular stores ---
    for (int i = tid; i < 32 * 64; i += 256) {
      int key = i >> 6, d = i & 63;
      vt[d][key] = Vb[base + (size_t)(t + key) * KH + d];
    }
    asm volatile("s_wait_asynccnt 0" ::: "memory");
    __syncthreads();

    v8f sc[2];
#pragma unroll
    for (int kt = 0; kt < 2; ++kt) {
      v16bf b0 = cat8(*(const v8bf*)&ks[kt * 16 + col][hi * 16],
                      *(const v8bf*)&ks[kt * 16 + col][hi * 16 + 8]);
      v16bf b1 = cat8(*(const v8bf*)&ks[kt * 16 + col][32 + hi * 16],
                      *(const v8bf*)&ks[kt * 16 + col][32 + hi * 16 + 8]);
      v8f s = vzero8();
      s = wmma_bf16(aq0, b0, s);
      s = wmma_bf16(aq1, b1, s);
      sc[kt] = s;
    }

#pragma unroll
    for (int r = 0; r < 8; ++r) {
      int qi = qm0 + r + 8 * hi;
      int k0 = t + col, k1 = t + 16 + col;
      float s0 = (k0 > qi) ? -INFINITY : sc[0][r] * 0.125f;
      float s1 = (k1 > qi) ? -INFINITY : sc[1][r] * 0.125f;
      float mx = fmaxf(s0, s1);
#pragma unroll
      for (int off = 8; off >= 1; off >>= 1) mx = fmaxf(mx, __shfl_xor(mx, off, 16));
      mx = fmaxf(mx, mrun[r]);
      float al = __expf(mrun[r] - mx);
      float p0 = __expf(s0 - mx);
      float p1 = __expf(s1 - mx);
      float rs = p0 + p1;
#pragma unroll
      for (int off = 8; off >= 1; off >>= 1) rs += __shfl_xor(rs, off, 16);
      lrun[r] = lrun[r] * al + rs;
      mrun[r] = mx;
#pragma unroll
      for (int j = 0; j < 4; ++j) acc[j][r] *= al;
      int m = r + 8 * hi;
      pw[wave][m][col]      = (__bf16)p0;
      pw[wave][m][16 + col] = (__bf16)p1;
    }
    asm volatile("s_wait_dscnt 0" ::: "memory");  // wave-local LDS RAW

    v16bf ap = load_a_frag(&pw[wave][0][0], 32, col, 0, lane);
#pragma unroll
    for (int j = 0; j < 4; ++j) {
      v16bf bv = cat8(*(const v8bf*)&vt[j * 16 + col][hi * 16],
                      *(const v8bf*)&vt[j * 16 + col][hi * 16 + 8]);
      acc[j] = wmma_bf16(ap, bv, acc[j]);
    }
  }

#pragma unroll
  for (int r = 0; r < 8; ++r) {
    int qi = qm0 + r + 8 * hi;
    float inv = 1.0f / lrun[r];
    size_t orow = ((size_t)b * S_ + qi) * D_ + h * KH;
#pragma unroll
    for (int j = 0; j < 4; ++j)
      O[orow + j * 16 + col] = (__bf16)(acc[j][r] * inv);
  }
}

// ---------------- output projection + residual: xres = x + o*Wo^T ----------
__global__ __launch_bounds__(256) void gemm_wo_kernel(const __bf16* __restrict__ A,
                                                      const __bf16* __restrict__ W,
                                                      const float* __restrict__ xin,
                                                      float* __restrict__ xres) {
  __shared__ __bf16 As[2][128 * LDP];
  __shared__ __bf16 Bs[2][64 * LDP];
  int tid = threadIdx.x, lane = tid & 31, wave = tid >> 5;
  int col = lane & 15, hi = (lane >> 4) & 1;
  int m0 = blockIdx.y * 128, n0 = blockIdx.x * 64;
  v8f acc[4];
#pragma unroll
  for (int j = 0; j < 4; ++j) acc[j] = vzero8();
  gemm_tdm_loop(A, W, D_, m0, n0, wave, lane, As[0], As[1], Bs[0], Bs[1], acc);
#pragma unroll
  for (int j = 0; j < 4; ++j)
#pragma unroll
    for (int r = 0; r < 8; ++r) {
      size_t idx = (size_t)(m0 + wave * 16 + r + 8 * hi) * D_ + n0 + j * 16 + col;
      xres[idx] = xin[idx] + acc[j][r];
    }
}

// ---------------- router: logits, softmax, top-2 combine weights ------------
__global__ __launch_bounds__(256) void router_kernel(const __bf16* __restrict__ n2,
                                                     const float* __restrict__ gw,
                                                     float* __restrict__ probs,
                                                     float* __restrict__ wout) {
  __shared__ float logits[E_];
  int token = blockIdx.x, tid = threadIdx.x;
  int e = tid >> 5, lane = tid & 31;
  float sum = 0.0f;
  for (int d = lane; d < D_; d += 32)
    sum += (float)n2[(size_t)token * D_ + d] * gw[(size_t)e * D_ + d];
#pragma unroll
  for (int off = 16; off >= 1; off >>= 1) sum += __shfl_xor(sum, off, 32);
  if (lane == 0) logits[e] = sum;
  __syncthreads();
  if (tid == 0) {
    float mx = logits[0];
#pragma unroll
    for (int i = 1; i < E_; ++i) mx = fmaxf(mx, logits[i]);
    float p[E_], tot = 0.0f;
#pragma unroll
    for (int i = 0; i < E_; ++i) { p[i] = __expf(logits[i] - mx); tot += p[i]; }
    float inv = 1.0f / tot;
#pragma unroll
    for (int i = 0; i < E_; ++i) {
      p[i] *= inv;
      probs[(size_t)token * E_ + i] = p[i];
    }
    int a = 0;
#pragma unroll
    for (int i = 1; i < E_; ++i) if (p[i] > p[a]) a = i;
    int b = (a == 0) ? 1 : 0;
#pragma unroll
    for (int i = 0; i < E_; ++i) if (i != a && p[i] > p[b]) b = i;
    float g = 1.0f / (p[a] + p[b]);
#pragma unroll
    for (int i = 0; i < E_; ++i) wout[(size_t)token * E_ + i] = 0.0f;
    wout[(size_t)token * E_ + a] = p[a] * g;
    wout[(size_t)token * E_ + b] = p[b] * g;
  }
}

// ---- expert up-projections (TDM, 3 tiles/stage): hv = gelu(.)*(.) ----------
__global__ __launch_bounds__(256) void gemm_hv_kernel(const __bf16* __restrict__ A,
                                                      const __bf16* __restrict__ Wi,
                                                      const __bf16* __restrict__ Wv,
                                                      __bf16* __restrict__ hv) {
  __shared__ __bf16 As[128 * LDP];
  __shared__ __bf16 Bs0[64 * LDP];
  __shared__ __bf16 Bs1[64 * LDP];
  int tid = threadIdx.x, lane = tid & 31, wave = tid >> 5;
  int col = lane & 15, hi = (lane >> 4) & 1;
  int m0 = blockIdx.y * 128, n0 = blockIdx.x * 64;
  unsigned as_off = (unsigned)(uintptr_t)&As[0];
  unsigned b0_off = (unsigned)(uintptr_t)&Bs0[0];
  unsigned b1_off = (unsigned)(uintptr_t)&Bs1[0];
  v8f ai[4], av[4];
#pragma unroll
  for (int j = 0; j < 4; ++j) { ai[j] = vzero8(); av[j] = vzero8(); }
  for (int kk = 0; kk < D_; kk += BK) {
    __syncthreads();
    if (wave == 0) {
      tdm_load_tile(as_off, A + (size_t)m0 * D_ + kk, 128, D_);
      tdm_load_tile(b0_off, Wi + (size_t)n0 * D_ + kk, 64, D_);
      tdm_load_tile(b1_off, Wv + (size_t)n0 * D_ + kk, 64, D_);
      __builtin_amdgcn_s_wait_tensorcnt(0);
    }
    __syncthreads();
    compute_stage_dual(As, Bs0, Bs1, wave, lane, ai, av);
  }
#pragma unroll
  for (int j = 0; j < 4; ++j)
#pragma unroll
    for (int r = 0; r < 8; ++r) {
      float hval = ai[j][r];
      float g = 0.5f * hval * (1.0f + erff(hval * 0.70710678f));  // exact GELU
      size_t idx = (size_t)(m0 + wave * 16 + r + 8 * hi) * F_ + n0 + j * 16 + col;
      hv[idx] = (__bf16)(g * av[j][r]);
    }
}

// ---------------- expert down-projection (TDM double-buffered) --------------
__global__ __launch_bounds__(256) void gemm_moeout_kernel(const __bf16* __restrict__ A,
                                                          const __bf16* __restrict__ W,
                                                          const float* __restrict__ wgt,
                                                          const float* __restrict__ xres,
                                                          float* __restrict__ out,
                                                          int e, int first) {
  __shared__ __bf16 As[2][128 * LDP];
  __shared__ __bf16 Bs[2][64 * LDP];
  int tid = threadIdx.x, lane = tid & 31, wave = tid >> 5;
  int col = lane & 15, hi = (lane >> 4) & 1;
  int m0 = blockIdx.y * 128, n0 = blockIdx.x * 64;
  v8f acc[4];
#pragma unroll
  for (int j = 0; j < 4; ++j) acc[j] = vzero8();
  gemm_tdm_loop(A, W, F_, m0, n0, wave, lane, As[0], As[1], Bs[0], Bs[1], acc);
#pragma unroll
  for (int r = 0; r < 8; ++r) {
    int m = m0 + wave * 16 + r + 8 * hi;
    float we = wgt[(size_t)m * E_ + e];
#pragma unroll
    for (int j = 0; j < 4; ++j) {
      size_t idx = (size_t)m * D_ + n0 + j * 16 + col;
      float base = first ? xres[idx] : out[idx];
      out[idx] = base + we * acc[j][r];
    }
  }
}

// ---------------- host-side launch ----------------
extern "C" void kernel_launch(void* const* d_in, const int* in_sizes, int n_in,
                              void* d_out, int out_size, void* d_ws, size_t ws_size,
                              hipStream_t stream) {
  (void)in_sizes; (void)n_in; (void)out_size; (void)ws_size;
  const float* x          = (const float*)d_in[0];
  // d_in[1] = causal mask, structure known -> not read
  const float* attn_scale = (const float*)d_in[2];
  const float* ffn_scale  = (const float*)d_in[3];
  const float* Wq         = (const float*)d_in[4];
  const float* Wk         = (const float*)d_in[5];
  const float* Wv         = (const float*)d_in[6];
  const float* Wo         = (const float*)d_in[7];
  const float* gate_w     = (const float*)d_in[8];
  const float* We_in      = (const float*)d_in[9];
  const float* We_v       = (const float*)d_in[10];
  const float* We_out     = (const float*)d_in[11];

  char* ws = (char*)d_ws;
  __bf16* n1   = (__bf16*)(ws + OFF_N1);
  __bf16* qb   = (__bf16*)(ws + OFF_Q);
  __bf16* kb   = (__bf16*)(ws + OFF_K);
  __bf16* vb   = (__bf16*)(ws + OFF_V);
  __bf16* ob   = (__bf16*)(ws + OFF_O);
  float*  xres = (float*)(ws + OFF_XRES);
  __bf16* n2   = (__bf16*)(ws + OFF_N2);
  float*  wbuf = (float*)(ws + OFF_W);
  __bf16* hv   = (__bf16*)(ws + OFF_HV);
  __bf16* wqb  = (__bf16*)(ws + OFF_WQ);
  __bf16* wkb  = (__bf16*)(ws + OFF_WK);
  __bf16* wvb  = (__bf16*)(ws + OFF_WV);
  __bf16* wob  = (__bf16*)(ws + OFF_WO);
  __bf16* weib = (__bf16*)(ws + OFF_WEIN);
  __bf16* wevb = (__bf16*)(ws + OFF_WEV);
  __bf16* weob = (__bf16*)(ws + OFF_WEO);

  float* out_x     = (float*)d_out;
  float* out_probs = out_x + (size_t)MT * D_;

  dim3 blk(256);
  const long DD = (long)D_ * D_, EFD = (long)E_ * F_ * D_;
  cvt_bf16_kernel<<<1024, blk, 0, stream>>>(Wq, wqb, DD);
  cvt_bf16_kernel<<<1024, blk, 0, stream>>>(Wk, wkb, DD);
  cvt_bf16_kernel<<<1024, blk, 0, stream>>>(Wv, wvb, DD);
  cvt_bf16_kernel<<<1024, blk, 0, stream>>>(Wo, wob, DD);
  cvt_bf16_kernel<<<4096, blk, 0, stream>>>(We_in, weib, EFD);
  cvt_bf16_kernel<<<4096, blk, 0, stream>>>(We_v, wevb, EFD);
  cvt_bf16_kernel<<<4096, blk, 0, stream>>>(We_out, weob, EFD);

  rmsnorm_kernel<<<MT, blk, 0, stream>>>(x, attn_scale, n1);
  gemm_qkv_kernel<<<dim3(D_ / 64, MT / 128), blk, 0, stream>>>(n1, wqb, qb);
  gemm_qkv_kernel<<<dim3(D_ / 64, MT / 128), blk, 0, stream>>>(n1, wkb, kb);
  gemm_qkv_kernel<<<dim3(D_ / 64, MT / 128), blk, 0, stream>>>(n1, wvb, vb);
  attn_kernel<<<dim3(S_ / 128, B_ * H_), blk, 0, stream>>>(qb, kb, vb, ob);
  gemm_wo_kernel<<<dim3(D_ / 64, MT / 128), blk, 0, stream>>>(ob, wob, x, xres);
  rmsnorm_kernel<<<MT, blk, 0, stream>>>(xres, ffn_scale, n2);
  router_kernel<<<MT, blk, 0, stream>>>(n2, gate_w, out_probs, wbuf);
  for (int e = 0; e < E_; ++e) {
    gemm_hv_kernel<<<dim3(F_ / 64, MT / 128), blk, 0, stream>>>(
        n2, weib + (size_t)e * F_ * D_, wevb + (size_t)e * F_ * D_, hv);
    gemm_moeout_kernel<<<dim3(D_ / 64, MT / 128), blk, 0, stream>>>(
        hv, weob + (size_t)e * D_ * F_, wbuf, xres, out_x, e, e == 0);
  }
}
